// CrossNetMix_73658689126796
// MI455X (gfx1250) — compile-verified, matrix operand order checked by reading, not run on hardware
//
#include <hip/hip_runtime.h>
#include <hip/hip_bf16.h>

typedef __bf16 bf16_t;
typedef bf16_t v16bf __attribute__((ext_vector_type(16)));
typedef bf16_t v8bf  __attribute__((ext_vector_type(8)));
typedef float  v8f   __attribute__((ext_vector_type(8)));

namespace {
constexpr int kBatch  = 16384;
constexpr int kDim    = 1024;
constexpr int kLayers = 3;
constexpr int kExp    = 4;
constexpr int kRank   = 64;
constexpr int kMTile  = 32;             // batch rows per workgroup
constexpr int kXLs    = 1036;           // padded f32 stride (bank-conflict-free, 16B row align)
constexpr int kVs     = 264;            // padded bf16 stride

// ---- pre-swizzled bf16 weight fragments in d_ws (element offsets, bf16 units) ----
constexpr int kWsV = 0;                            // (layer*32+kk)*16+nt : 1536 frags
constexpr int kWsU = kWsV + 1536 * 512;            // (layer*8+kk)*64+nt  : 1536 frags
constexpr int kWsC = kWsU + 1536 * 512;            // ((layer*4+e)*8+ntl*2+kk) : 96 frags
constexpr int kWsG = kWsC + 96 * 512;              // kk : 32 frags (cols>=4 zero)

constexpr int kUnitsV = 1536 * 32;                 // one unit = 16 bf16 (one lane of a frag)
constexpr int kUnitsU = 1536 * 32;
constexpr int kUnitsC = 96 * 32;
constexpr int kUnitsG = 32 * 32;
constexpr int kUnits  = kUnitsV + kUnitsU + kUnitsC + kUnitsG;   // 102400
}

__device__ __forceinline__ float fast_tanh(float x) {
    const float ax = __builtin_fabsf(x);
    const float t  = __expf(-2.0f * ax);                       // v_exp_f32, branch-free
    const float y  = (1.0f - t) * __builtin_amdgcn_rcpf(1.0f + t);
    return __builtin_copysignf(y, x);
}

__device__ __forceinline__ v8f wmma_bf16(v16bf a, v16bf b, v8f c) {
    return __builtin_amdgcn_wmma_f32_16x16x32_bf16(false, a, false, b, (short)0, c, false, false);
}

// A-fragment (16x32 bf16) from f32 LDS row: two runs of 8 consecutive floats.
__device__ __forceinline__ v16bf load_a_f32(const float* p) {
    const float4 f0 = *reinterpret_cast<const float4*>(p);
    const float4 f1 = *reinterpret_cast<const float4*>(p + 4);
    const float4 f2 = *reinterpret_cast<const float4*>(p + 16);
    const float4 f3 = *reinterpret_cast<const float4*>(p + 20);
    v16bf a;
    a[0]=(bf16_t)f0.x; a[1]=(bf16_t)f0.y; a[2]=(bf16_t)f0.z; a[3]=(bf16_t)f0.w;
    a[4]=(bf16_t)f1.x; a[5]=(bf16_t)f1.y; a[6]=(bf16_t)f1.z; a[7]=(bf16_t)f1.w;
    a[8]=(bf16_t)f2.x; a[9]=(bf16_t)f2.y; a[10]=(bf16_t)f2.z; a[11]=(bf16_t)f2.w;
    a[12]=(bf16_t)f3.x; a[13]=(bf16_t)f3.y; a[14]=(bf16_t)f3.z; a[15]=(bf16_t)f3.w;
    return a;
}

// A-fragment from bf16 LDS row: two 16-byte runs -> 2x ds_load_b128, no cvt.
__device__ __forceinline__ v16bf load_a_bf16(const bf16_t* p) {
    const v8bf lo = *reinterpret_cast<const v8bf*>(p);
    const v8bf hi = *reinterpret_cast<const v8bf*>(p + 16);
    v16bf a;
#pragma unroll
    for (int i = 0; i < 8; ++i) { a[i] = lo[i]; a[i + 8] = hi[i]; }
    return a;
}

// ---------------- weight pre-swizzle: f32 tensors -> bf16 WMMA-B fragments ----------------
__global__ __launch_bounds__(256)
void swizzle_weights(const float* __restrict__ U, const float* __restrict__ V,
                     const float* __restrict__ C, const float* __restrict__ G,
                     bf16_t* __restrict__ ws)
{
    const int u = blockIdx.x * 256 + threadIdx.x;
    if (u >= kUnits) return;
    const int lane = u & 31;
    const int kHi  = (lane < 16) ? 0 : 8;
    const int c    = lane & 15;
    v16bf vals;
    bf16_t* dst;
    if (u < kUnitsV) {
        const int frag = u >> 5;                 // (l*32+kk)*16+nt
        const int l  = frag >> 9;
        const int kk = (frag >> 4) & 31;
        const int nt = frag & 15;
        const int col = nt * 16 + c;
        const int e = col >> 6, r = col & 63;
        const float* src = V + ((size_t)(l * kExp + e) * kDim) * kRank + r;
#pragma unroll
        for (int i = 0; i < 16; ++i) {
            const int k = kk * 32 + ((i < 8) ? i : (i + 8)) + kHi;
            vals[i] = (bf16_t)src[(size_t)k * kRank];
        }
        dst = ws + kWsV + (size_t)frag * 512 + lane * 16;
    } else if (u < kUnitsV + kUnitsU) {
        const int frag = (u - kUnitsV) >> 5;     // (l*8+kk)*64+nt
        const int l  = frag >> 9;
        const int kk = (frag >> 6) & 7;
        const int nt = frag & 63;
        const int d = nt * 16 + c;
#pragma unroll
        for (int i = 0; i < 16; ++i) {
            const int k = kk * 32 + ((i < 8) ? i : (i + 8)) + kHi;
            const int e = k >> 6, r = k & 63;
            vals[i] = (bf16_t)U[((size_t)(l * kExp + e) * kDim + d) * kRank + r];
        }
        dst = ws + kWsU + (size_t)frag * 512 + lane * 16;
    } else if (u < kUnitsV + kUnitsU + kUnitsC) {
        const int frag = (u - kUnitsV - kUnitsU) >> 5;   // ((l*4+e)*8+ntl*2+kk)
        const int l   = frag >> 5;
        const int e   = (frag >> 3) & 3;
        const int ntl = (frag >> 1) & 3;
        const int kk  = frag & 1;
        const int r   = ntl * 16 + c;
        const float* src = C + ((size_t)(l * kExp + e) * kRank + r) * kRank;
#pragma unroll
        for (int i = 0; i < 16; ++i) {
            const int s = kk * 32 + ((i < 8) ? i : (i + 8)) + kHi;
            vals[i] = (bf16_t)src[s];
        }
        dst = ws + kWsC + (size_t)frag * 512 + lane * 16;
    } else {
        const int frag = (u - kUnitsV - kUnitsU - kUnitsC) >> 5;  // kk
#pragma unroll
        for (int i = 0; i < 16; ++i) {
            const int k = frag * 32 + ((i < 8) ? i : (i + 8)) + kHi;
            vals[i] = (c < kExp) ? (bf16_t)G[(size_t)c * kDim + k] : (bf16_t)0.0f;
        }
        dst = ws + kWsG + (size_t)frag * 512 + lane * 16;
    }
    *reinterpret_cast<v16bf*>(dst) = vals;
}

// ---------------- fused 3-layer CrossNetMix ----------------
__global__ __launch_bounds__(256, 2)
void crossnet_mix_fused(const float* __restrict__ x,
                        const float* __restrict__ bias,
                        const bf16_t* __restrict__ ws,
                        float* __restrict__ out)
{
    __shared__ float  xl[kMTile * kXLs];      // resident x_l tile (f32)
    __shared__ bf16_t vb[kMTile * kVs];       // v, then gate-scaled c~ (bf16)
    __shared__ float  gateb[kMTile * kExp];

    const int tid  = threadIdx.x;
    const int lane = tid & 31;
    const int wave = tid >> 5;
    const int rowA = lane & 15;
    const int kHi  = (lane < 16) ? 0 : 8;
    const int rowBase = blockIdx.x * kMTile;

    // stage x tile -> xl
    for (int f = tid; f < kMTile * (kDim / 4); f += 256) {
        const int b = f >> 8, c4 = f & 255;
        const float4 val = reinterpret_cast<const float4*>(x + (size_t)(rowBase + b) * kDim)[c4];
        *reinterpret_cast<float4*>(&xl[b * kXLs + c4 * 4]) = val;
    }
    __syncthreads();

    const int mt = wave & 1;        // row tile owned by this wave
    const int ng = wave >> 1;       // 0..3 column group

    for (int layer = 0; layer < kLayers; ++layer) {
        const float* bl = bias + (size_t)layer * kDim;

        // ---- gate logits via WMMA: xl(16x1024) @ Gpad(1024x16); waves 0,1 only ----
        if (wave < 2) {
            const float*  ap = &xl[(wave * 16 + rowA) * kXLs] + kHi;
            const bf16_t* gp = ws + kWsG + lane * 16;
            v8f acc = {0.f,0.f,0.f,0.f,0.f,0.f,0.f,0.f};
#pragma unroll 2
            for (int kk = 0; kk < 32; ++kk) {
                const v16bf b = *reinterpret_cast<const v16bf*>(gp);
                const v16bf a = load_a_f32(ap);
                acc = wmma_bf16(a, b, acc);
                gp += 512; ap += 32;
            }
            if (rowA < kExp) {
#pragma unroll
                for (int j = 0; j < 8; ++j)
                    gateb[(wave * 16 + j + kHi) * kExp + rowA] = acc[j];
            }
        }

        // ---- GEMM1: v = tanh(xl(32x1024) @ V_all(1024x256)); 1x4 blocking, batched loads ----
        {
            const float*  ap = &xl[(mt * 16 + rowA) * kXLs] + kHi;
            const bf16_t* wp = ws + kWsV + ((size_t)(layer * 32) * 16 + ng * 4) * 512 + lane * 16;
            v8f acc[4];
#pragma unroll
            for (int q = 0; q < 4; ++q) acc[q] = (v8f){0.f,0.f,0.f,0.f,0.f,0.f,0.f,0.f};
#pragma unroll 2
            for (int kk = 0; kk < 32; ++kk) {
                const v16bf b0 = *reinterpret_cast<const v16bf*>(wp + 0 * 512);
                const v16bf b1 = *reinterpret_cast<const v16bf*>(wp + 1 * 512);
                const v16bf b2 = *reinterpret_cast<const v16bf*>(wp + 2 * 512);
                const v16bf b3 = *reinterpret_cast<const v16bf*>(wp + 3 * 512);
                const v16bf a  = load_a_f32(ap);
                acc[0] = wmma_bf16(a, b0, acc[0]);
                acc[1] = wmma_bf16(a, b1, acc[1]);
                acc[2] = wmma_bf16(a, b2, acc[2]);
                acc[3] = wmma_bf16(a, b3, acc[3]);
                wp += 16 * 512; ap += 32;
            }
#pragma unroll
            for (int q = 0; q < 4; ++q) {
                const int col = (ng * 4 + q) * 16 + rowA;
#pragma unroll
                for (int j = 0; j < 8; ++j) {
                    const int row = mt * 16 + j + kHi;
                    vb[row * kVs + col] = (bf16_t)fast_tanh(acc[q][j]);
                }
            }
        }
        __syncthreads();

        // softmax over E=4 (wave 0 threads)
        if (tid < kMTile) {
            float g0 = gateb[tid*4+0], g1 = gateb[tid*4+1], g2 = gateb[tid*4+2], g3 = gateb[tid*4+3];
            float m  = fmaxf(fmaxf(g0, g1), fmaxf(g2, g3));
            float e0 = __expf(g0-m), e1 = __expf(g1-m), e2 = __expf(g2-m), e3 = __expf(g3-m);
            float inv = __builtin_amdgcn_rcpf(e0 + e1 + e2 + e3);
            gateb[tid*4+0]=e0*inv; gateb[tid*4+1]=e1*inv; gateb[tid*4+2]=e2*inv; gateb[tid*4+3]=e3*inv;
        }

        // ---- GEMM2: c_e = v_e(32x64) @ C_e^T(64x64) per expert; batched loads ----
        float creg[4][8];
#pragma unroll
        for (int it = 0; it < 4; ++it) {
            const int t = wave + it * 8;
            const int e = t >> 3, mt2 = (t >> 2) & 1, ntl = t & 3;
            const bf16_t* vrow = &vb[(mt2 * 16 + rowA) * kVs + e * 64];
            const bf16_t* cp = ws + kWsC + (size_t)((layer * kExp + e) * 8 + ntl * 2) * 512 + lane * 16;
            const v16bf c0 = *reinterpret_cast<const v16bf*>(cp);
            const v16bf c1 = *reinterpret_cast<const v16bf*>(cp + 512);
            const v16bf a0 = load_a_bf16(vrow + kHi);
            const v16bf a1 = load_a_bf16(vrow + 32 + kHi);
            v8f acc = {0.f,0.f,0.f,0.f,0.f,0.f,0.f,0.f};
            acc = wmma_bf16(a0, c0, acc);
            acc = wmma_bf16(a1, c1, acc);
#pragma unroll
            for (int j = 0; j < 8; ++j) creg[it][j] = acc[j];
        }
        __syncthreads();   // vb reads done + softmax done

        // ---- c~ = gate * tanh(c) -> vb ----
#pragma unroll
        for (int it = 0; it < 4; ++it) {
            const int t = wave + it * 8;
            const int e = t >> 3, mt2 = (t >> 2) & 1, ntl = t & 3;
            const int col = e * 64 + ntl * 16 + rowA;
#pragma unroll
            for (int j = 0; j < 8; ++j) {
                const int row = mt2 * 16 + j + kHi;
                vb[row * kVs + col] = (bf16_t)(gateb[row * kExp + e] * fast_tanh(creg[it][j]));
            }
        }
        __syncthreads();

        // ---- GEMM3: u~ = c~(32x256) @ U_all(256x1024); x_l += x0*(u~+bias) ----
        const bf16_t* vrow = &vb[(mt * 16 + rowA) * kVs];
#pragma unroll
        for (int pass = 0; pass < 2; ++pass) {
            v8f acc[8];
#pragma unroll
            for (int q = 0; q < 8; ++q) acc[q] = (v8f){0.f,0.f,0.f,0.f,0.f,0.f,0.f,0.f};
            const bf16_t* vp = vrow + kHi;
            const bf16_t* wp = ws + kWsU +
                ((size_t)(layer * 8) * 64 + ng * 16 + pass * 8) * 512 + lane * 16;
#pragma unroll
            for (int kk = 0; kk < 8; ++kk) {       // fully unrolled: immediates + pipelining
                v16bf b[8];
#pragma unroll
                for (int q = 0; q < 8; ++q)
                    b[q] = *reinterpret_cast<const v16bf*>(wp + (size_t)(kk * 64 + q) * 512);
                const v16bf a = load_a_bf16(vp + kk * 32);
#pragma unroll
                for (int q = 0; q < 8; ++q) acc[q] = wmma_bf16(a, b[q], acc[q]);
            }
            const float* xrow = x + (size_t)rowBase * kDim;
#pragma unroll
            for (int q = 0; q < 8; ++q) {
                const int d = (ng * 16 + pass * 8 + q) * 16 + rowA;
                const float bv = bl[d];
#pragma unroll
                for (int j = 0; j < 8; ++j) {
                    const int row = mt * 16 + j + kHi;
                    const float x0v = xrow[(size_t)row * kDim + d];
                    xl[row * kXLs + d] += x0v * (acc[q][j] + bv);
                }
            }
        }
        __syncthreads();
    }

    // write x_l tile to output
    for (int f = tid; f < kMTile * (kDim / 4); f += 256) {
        const int b = f >> 8, c4 = f & 255;
        const float4 val = *reinterpret_cast<const float4*>(&xl[b * kXLs + c4 * 4]);
        reinterpret_cast<float4*>(out + (size_t)(rowBase + b) * kDim)[c4] = val;
    }
}

extern "C" void kernel_launch(void* const* d_in, const int* in_sizes, int n_in,
                              void* d_out, int out_size, void* d_ws, size_t ws_size,
                              hipStream_t stream) {
    (void)in_sizes; (void)n_in; (void)out_size; (void)ws_size;
    const float* x    = (const float*)d_in[0];
    const float* U    = (const float*)d_in[1];
    const float* V    = (const float*)d_in[2];
    const float* C    = (const float*)d_in[3];
    const float* bias = (const float*)d_in[4];
    const float* G    = (const float*)d_in[5];
    bf16_t* ws = (bf16_t*)d_ws;
    float* out = (float*)d_out;

    swizzle_weights<<<dim3((kUnits + 255) / 256), dim3(256), 0, stream>>>(U, V, C, G, ws);
    crossnet_mix_fused<<<dim3(kBatch / kMTile), dim3(256), 0, stream>>>(x, bias, ws, out);
}
